// Morpher_21105469292862
// MI455X (gfx1250) — matrix operand order repaired and checked by reading, not compiled, not run on hardware
//
#include <hip/hip_runtime.h>
#include <hip/hip_bf16.h>

#define B_  4
#define T_  1024
#define K_  3
#define N_  4
#define D_  64
#define H_  (K_ * N_)      // 12
#define DO_ (3 * D_)       // 192

typedef __attribute__((ext_vector_type(16))) __bf16 v16bf;
typedef __attribute__((ext_vector_type(8)))  float  v8f;

__device__ __forceinline__ unsigned short f2bf(float f) {
    unsigned int u = __float_as_uint(f);
    u += 0x7FFFu + ((u >> 16) & 1u);     // round-to-nearest-even
    return (unsigned short)(u >> 16);
}

union BfPack { uint4 q[2]; v16bf v; };

// A operand: 16(M) x 32(K) bf16 tile from row-major [rows][C] bf16 LDS (ISA 7.12.2).
__device__ __forceinline__ v16bf load_a(const unsigned short* base, int C, int row0, int kb) {
    int lane = threadIdx.x & 31;
    int off0 = (row0 + (lane & 15)) * C + kb + (lane >> 4) * 8;
    BfPack p;
    p.q[0] = *(const uint4*)(base + off0);
    p.q[1] = *(const uint4*)(base + off0 + 16);
    return p.v;
}

// B operand: 32(K) x 16(N); storage [ncol][Ck] with contraction contiguous.
__device__ __forceinline__ v16bf load_b(const unsigned short* base, int C, int ncol0, int kb) {
    int lane = threadIdx.x & 31;
    int off = (ncol0 + (lane & 15)) * C + kb + (lane >> 4) * 8;
    BfPack p;
    p.q[0] = *(const uint4*)(base + off);
    p.q[1] = *(const uint4*)(base + off + 16);
    return p.v;
}

__device__ __forceinline__ v8f wmma_bf16(v16bf a, v16bf b, v8f c) {
    return __builtin_amdgcn_wmma_f32_16x16x32_bf16(false, a, false, b, (short)0, c, false, false);
}

// CDNA5 async copy: per-lane 16B global -> LDS, tracked by ASYNCcnt.
__device__ __forceinline__ void async_ld_b128(unsigned int lds_byte_off, const void* gptr) {
    asm volatile("global_load_async_to_lds_b128 %0, %1, off"
                 :: "v"(lds_byte_off), "v"(gptr)
                 : "memory");
}
__device__ __forceinline__ void wait_async() {
    asm volatile("s_wait_asynccnt 0" ::: "memory");
}

#define QSCL (0.125f * 1.44269504088896340736f)   // (1/sqrt(D)) * log2(e)

// ---------------------------------------------------------------------------
// Kernel 1: qkv[b,t,k,n,:] = x[b,t,k,n,:] @ Wqkv[k, s2h[k,n], :, :]
// Q pre-scaled by log2(e)/sqrt(D); Q/K/V stored bf16 [B*H, T, D] in workspace.
// ---------------------------------------------------------------------------
__global__ void __launch_bounds__(256) qkv_proj_kernel(
    const float* __restrict__ x, const float* __restrict__ Wqkv,
    const int* __restrict__ s2h,
    unsigned short* __restrict__ Qws, unsigned short* __restrict__ Kws,
    unsigned short* __restrict__ Vws)
{
    __shared__ __align__(16) unsigned short sX[128 * 64];   // [t][i] bf16
    __shared__ __align__(16) unsigned short sW[192 * 64];   // transposed [o][i] bf16

    int bh = blockIdx.x;
    int b = bh / H_, h = bh % H_;
    int k = h / N_, n = h % N_;
    int t0 = blockIdx.y * 128;
    int tid = threadIdx.x;

    int hsel = s2h[k * N_ + n];
    const float* Wp = Wqkv + (size_t)(k * N_ + hsel) * D_ * DO_;
    const float* Xp = x + ((size_t)((b * T_ + t0) * K_ + k) * N_ + n) * D_;

    // X tile: float4 reads, packed bf16x2 LDS stores (ds_store_b64)
    for (int idx = tid; idx < 128 * 16; idx += 256) {
        int r = idx >> 4, c4 = idx & 15;
        float4 f = *(const float4*)(Xp + (size_t)r * (K_ * N_ * D_) + c4 * 4);
        unsigned int u0 = (unsigned)f2bf(f.x) | ((unsigned)f2bf(f.y) << 16);
        unsigned int u1 = (unsigned)f2bf(f.z) | ((unsigned)f2bf(f.w) << 16);
        *(uint2*)&sX[r * 64 + c4 * 4] = make_uint2(u0, u1);
    }
    // W: float4 reads along o, transposed bf16 stores
    for (int idx = tid; idx < 64 * 48; idx += 256) {
        int i = idx / 48, o4 = idx % 48;
        float4 f = *(const float4*)(Wp + i * DO_ + o4 * 4);
        int o = o4 * 4;
        sW[(o + 0) * 64 + i] = f2bf(f.x);
        sW[(o + 1) * 64 + i] = f2bf(f.y);
        sW[(o + 2) * 64 + i] = f2bf(f.z);
        sW[(o + 3) * 64 + i] = f2bf(f.w);
    }
    __syncthreads();

    int wave = tid >> 5;
    int lane = tid & 31;
    int lh   = lane >> 4;
    int nlo  = lane & 15;
    int row0 = wave * 16;

    v16bf a0 = load_a(sX, 64, row0, 0);
    v16bf a1 = load_a(sX, 64, row0, 32);

    size_t obase = (size_t)(b * H_ + h) * T_ * D_;

    #pragma unroll 4
    for (int nt = 0; nt < 12; ++nt) {
        v8f acc = (v8f){};
        acc = wmma_bf16(a0, load_b(sW, 64, nt * 16, 0),  acc);
        acc = wmma_bf16(a1, load_b(sW, 64, nt * 16, 32), acc);
        float scl = (nt < 4) ? QSCL : 1.0f;
        unsigned short* dst = (nt < 4) ? Qws : (nt < 8 ? Kws : Vws);
        int d0 = (nt & 3) * 16 + nlo;
        #pragma unroll
        for (int r = 0; r < 8; ++r) {
            int row = t0 + row0 + r + lh * 8;
            dst[obase + (size_t)row * D_ + d0] = f2bf(acc[r] * scl);
        }
    }
}

// ---------------------------------------------------------------------------
// Kernel 2: causal flash attention, 128-row Q tiles, 8 waves/block.
// K & Q tiles filled via global_load_async_to_lds_b128 (ASYNCcnt), V
// transposed at fill, next tiles prefetched. exp2-domain online softmax.
// ---------------------------------------------------------------------------
__global__ void __launch_bounds__(256) flash_attn_kernel(
    const unsigned short* __restrict__ Qws, const unsigned short* __restrict__ Kws,
    const unsigned short* __restrict__ Vws, float* __restrict__ out)
{
    __shared__ __align__(16) unsigned short sQ[128 * 64];      // [t][d]
    __shared__ __align__(16) unsigned short sK[64 * 64];       // [s][d]
    __shared__ __align__(16) unsigned short sV[64 * 64];       // transposed [d][s]
    __shared__ __align__(16) unsigned short sP[8 * 16 * 64];   // per-wave P tiles

    int bh = blockIdx.x;
    int b = bh / H_, h = bh % H_;
    int qi = blockIdx.y;
    int q0 = qi * 128;
    int tid  = threadIdx.x;
    int wave = tid >> 5;
    int lane = tid & 31;
    int lh   = lane >> 4;
    int nlo  = lane & 15;

    size_t base = (size_t)bh * T_ * D_;
    unsigned int sQ_base = (unsigned int)(size_t)(void*)sQ;
    unsigned int sK_base = (unsigned int)(size_t)(void*)sK;

    // Q tile: async copy, 8192 halves = 1024 lane-b128 ops
    const unsigned short* Qp = Qws + base + (size_t)q0 * D_;
    #pragma unroll
    for (int it = 0; it < 4; ++it) {
        int e = (it * 256 + tid) * 8;
        async_ld_b128(sQ_base + e * 2, Qp + e);
    }
    wait_async();
    __syncthreads();

    v16bf qa0 = load_a(sQ, 64, wave * 16, 0);
    v16bf qa1 = load_a(sQ, 64, wave * 16, 32);

    float m_s[8], l_s[8];
    v8f O[4];
    #pragma unroll
    for (int r = 0; r < 8; ++r) { m_s[r] = -3.0e38f; l_s[r] = 0.0f; }
    #pragma unroll
    for (int nt = 0; nt < 4; ++nt) O[nt] = (v8f){};

    int jmax = 2 * qi + 1;
    for (int j = 0; j <= jmax; ++j) {
        int s0 = j * 64;
        const unsigned short* Kp = Kws + base + (size_t)s0 * D_;
        const unsigned short* Vp = Vws + base + (size_t)s0 * D_;

        __syncthreads();   // previous iteration done with sK/sV/sP
        // K tile: async global->LDS (bf16 copy, no conversion needed)
        #pragma unroll
        for (int it = 0; it < 2; ++it) {
            int e = (it * 256 + tid) * 8;
            async_ld_b128(sK_base + e * 2, Kp + e);
        }
        // V tile: uint4 reads, transposed bf16 stores
        #pragma unroll
        for (int it = 0; it < 2; ++it) {
            int e = it * 256 + tid;               // 512 groups of 8 halves
            int s = e >> 3, d0 = (e & 7) * 8;
            uint4 v = *(const uint4*)(Vp + s * 64 + d0);
            unsigned int w0 = v.x, w1 = v.y, w2 = v.z, w3 = v.w;
            sV[(d0 + 0) * 64 + s] = (unsigned short)(w0 & 0xFFFF);
            sV[(d0 + 1) * 64 + s] = (unsigned short)(w0 >> 16);
            sV[(d0 + 2) * 64 + s] = (unsigned short)(w1 & 0xFFFF);
            sV[(d0 + 3) * 64 + s] = (unsigned short)(w1 >> 16);
            sV[(d0 + 4) * 64 + s] = (unsigned short)(w2 & 0xFFFF);
            sV[(d0 + 5) * 64 + s] = (unsigned short)(w2 >> 16);
            sV[(d0 + 6) * 64 + s] = (unsigned short)(w3 & 0xFFFF);
            sV[(d0 + 7) * 64 + s] = (unsigned short)(w3 >> 16);
        }
        // prefetch next K/V tile into L2 while we compute this one
        if (j < jmax) {
            __builtin_prefetch(Kp + 4096 + (tid & 63) * 64, 0, 1);
            __builtin_prefetch(Vp + 4096 + (tid & 63) * 64, 0, 1);
        }
        wait_async();
        __syncthreads();

        // wave-uniform predicate: skip tiles entirely in the causal future
        bool act = (s0 <= q0 + wave * 16 + 15);
        if (act) {
            v8f S[4];
            #pragma unroll
            for (int nt = 0; nt < 4; ++nt) {
                v8f acc = (v8f){};
                acc = wmma_bf16(qa0, load_b(sK, 64, nt * 16, 0),  acc);
                acc = wmma_bf16(qa1, load_b(sK, 64, nt * 16, 32), acc);
                S[nt] = acc;
            }
            if (s0 + 63 > q0 + wave * 16) {    // diagonal: apply causal mask
                #pragma unroll
                for (int nt = 0; nt < 4; ++nt) {
                    int col = s0 + nt * 16 + nlo;
                    #pragma unroll
                    for (int r = 0; r < 8; ++r) {
                        int row = q0 + wave * 16 + r + lh * 8;
                        if (col > row) S[nt][r] = -3.0e38f;
                    }
                }
            }
            // online softmax (base-2 domain; Q carries log2e/sqrt(D))
            float a_s[8];
            #pragma unroll
            for (int r = 0; r < 8; ++r) {
                float rm = fmaxf(fmaxf(S[0][r], S[1][r]), fmaxf(S[2][r], S[3][r]));
                rm = fmaxf(rm, __shfl_xor(rm, 1, 32));
                rm = fmaxf(rm, __shfl_xor(rm, 2, 32));
                rm = fmaxf(rm, __shfl_xor(rm, 4, 32));
                rm = fmaxf(rm, __shfl_xor(rm, 8, 32));
                float m_new = fmaxf(m_s[r], rm);
                float alpha = exp2f(m_s[r] - m_new);
                float rs = 0.0f;
                #pragma unroll
                for (int nt = 0; nt < 4; ++nt) {
                    float p = exp2f(S[nt][r] - m_new);
                    S[nt][r] = p;
                    rs += p;
                }
                rs += __shfl_xor(rs, 1, 32);
                rs += __shfl_xor(rs, 2, 32);
                rs += __shfl_xor(rs, 4, 32);
                rs += __shfl_xor(rs, 8, 32);
                l_s[r] = l_s[r] * alpha + rs;
                m_s[r] = m_new;
                a_s[r] = alpha;
            }
            #pragma unroll
            for (int nt = 0; nt < 4; ++nt) {
                int col = nt * 16 + nlo;
                #pragma unroll
                for (int r = 0; r < 8; ++r) {
                    O[nt][r] *= a_s[r];
                    sP[wave * 1024 + (r + lh * 8) * 64 + col] = f2bf(S[nt][r]);
                }
            }
        }
        __syncthreads();   // P visible cross-lane; keeps all waves convergent
        if (act) {
            v16bf pa0 = load_a(sP + wave * 1024, 64, 0, 0);
            v16bf pa1 = load_a(sP + wave * 1024, 64, 0, 32);
            #pragma unroll
            for (int nt = 0; nt < 4; ++nt) {
                O[nt] = wmma_bf16(pa0, load_b(sV, 64, nt * 16, 0),  O[nt]);
                O[nt] = wmma_bf16(pa1, load_b(sV, 64, nt * 16, 32), O[nt]);
            }
        }
    }

    // normalize + store: out[b,t,k,n,d] == out[b][t][h][d]
    #pragma unroll
    for (int r = 0; r < 8; ++r) {
        float inv = 1.0f / l_s[r];
        int row = q0 + wave * 16 + r + lh * 8;
        #pragma unroll
        for (int nt = 0; nt < 4; ++nt) {
            int d = nt * 16 + nlo;
            out[((size_t)(b * T_ + row) * H_ + h) * D_ + d] = O[nt][r] * inv;
        }
    }
}

extern "C" void kernel_launch(void* const* d_in, const int* in_sizes, int n_in,
                              void* d_out, int out_size, void* d_ws, size_t ws_size,
                              hipStream_t stream) {
    (void)in_sizes; (void)n_in; (void)out_size; (void)ws_size;
    const float* x    = (const float*)d_in[0];
    const float* Wqkv = (const float*)d_in[1];
    const int*   s2h  = (const int*)d_in[2];
    float* out = (float*)d_out;

    size_t per = (size_t)B_ * H_ * T_ * D_;          // elements per Q/K/V plane
    unsigned short* Qws = (unsigned short*)d_ws;     // bf16 workspace: ~18.9 MB
    unsigned short* Kws = Qws + per;
    unsigned short* Vws = Kws + per;

    qkv_proj_kernel<<<dim3(B_ * H_, T_ / 128), 256, 0, stream>>>(x, Wqkv, s2h, Qws, Kws, Vws);
    flash_attn_kernel<<<dim3(B_ * H_, T_ / 64 / 2), 256, 0, stream>>>(Qws, Kws, Vws, out);
}
